// AnchorSelector_70334384439468
// MI455X (gfx1250) — matrix operand order repaired and checked by reading, not compiled, not run on hardware
//
#include <hip/hip_runtime.h>
#include <cstdint>
#include <cstddef>

// ---------------- problem constants ----------------
#define NUM_MAPS      5
#define BATCH         8
#define CCH           256
#define NCA           9              // cell anchors
#define KTOT          2304           // 256 * 9 (im2col K)
#define POS_PER_IMG   21824          // 128^2+64^2+32^2+16^2+8^2
#define ANCH_PER_IMG  196416        // POS_PER_IMG * 9
#define RELK          300
#define MTILE         32             // positions per conv block

__device__ __constant__ int d_cum[6] = {0, 16384, 20480, 21504, 21760, 21824};

typedef __attribute__((ext_vector_type(16))) __bf16 v16bf;
typedef __attribute__((ext_vector_type(8)))  float  v8f;

union FragBF { uint4 q[2]; v16bf v; };

__device__ inline unsigned short f2bf(float f) {
  union { float f; unsigned int u; } x; x.f = f;
  unsigned int u = x.u;
  u += 0x7FFFu + ((u >> 16) & 1u);       // round-to-nearest-even
  return (unsigned short)(u >> 16);
}

__device__ inline v8f wmma_bf16(v16bf a, v16bf b, v8f c) {
  return __builtin_amdgcn_wmma_f32_16x16x32_bf16(false, a, false, b, (short)0, c, false, false);
}

// ---------------- kernel 1: weight conversion ----------------
// wpre2[n][tap*256+c] = bf16(w_pre[n][c][tap])   (tap-major K so a K-block has uniform (ky,kx))
// wpostT[cell][e][d]  = bf16(w_post[cell][d][e]) (transposed: contiguous B-fragment loads)
__global__ __launch_bounds__(256)
void convert_weights_kernel(const float* __restrict__ w_pre, const float* __restrict__ w_post,
                            unsigned short* __restrict__ wpre2,
                            unsigned short* __restrict__ wpostT) {
  int i = blockIdx.x * 256 + threadIdx.x;
  if (i < CCH * KTOT) {                 // 589824 == 9*65536, reuse index for both arrays
    int n   = i / KTOT;
    int rem = i - n * KTOT;
    int tap = rem >> 8, c = rem & 255;
    wpre2[i] = f2bf(w_pre[(n * CCH + c) * 9 + tap]);
    int cell = i >> 16;
    int r2   = i & 65535;
    int e = r2 >> 8, d = r2 & 255;
    wpostT[(cell << 16) + (e << 8) + d] = f2bf(w_post[(cell << 16) + (d << 8) + e]);
  }
}

// ---------------- kernel 1b: feature prepass ----------------
// fmT[map][b][pos][c] = bf16(fm[map][b][c][pos])  -> every position is a contiguous 512B line
__global__ __launch_bounds__(256)
void prepass_kernel(const float* __restrict__ fm0, const float* __restrict__ fm1,
                    const float* __restrict__ fm2, const float* __restrict__ fm3,
                    const float* __restrict__ fm4,
                    unsigned short* __restrict__ fmT) {
  int id = blockIdx.x * 256 + threadIdx.x;          // [0, 8*21824*256)
  if (id >= BATCH * POS_PER_IMG * CCH) return;
  int c = id & 255;
  int r = id >> 8;
  int b = r / POS_PER_IMG;
  int q = r - b * POS_PER_IMG;
  int map = 0;
  while (map < 4 && q >= d_cum[map + 1]) ++map;
  int s = 128 >> map;
  int loc = q - d_cum[map];
  const float* fmp = (map == 0) ? fm0 : (map == 1) ? fm1 : (map == 2) ? fm2 : (map == 3) ? fm3 : fm4;
  float v = fmp[(b * CCH + c) * (s * s) + loc];
  fmT[2048 * d_cum[map] + (b * (s * s) + loc) * CCH + c] = f2bf(v);
}

// ---------------- kernel 2: fused conv3x3 + bias + ReLU + 1x1 proj ----------------
// Block = 32 positions x 256 out channels, 256 threads (8 waves).
// Wave w: N-tiles {w, w+8} x M-tiles {0,1}; B fragments reused across both M-tiles.
// K order: 9 taps x 8 channel-blocks of 32; per tap the A slab [32 x 256] bf16 is staged
// with coalesced b128 copies (uniform border test per row) then 8 WMMA K-steps run on it.
__global__ __launch_bounds__(256)
void conv_proj_kernel(const unsigned short* __restrict__ fmT,
                      const unsigned short* __restrict__ wpre2,
                      const float* __restrict__ b_pre,
                      const float* __restrict__ w_proj, const float* __restrict__ b_proj,
                      float* __restrict__ logits) {
  __shared__ __align__(16) unsigned short smT[MTILE * 264];  // tap slab, padded stride
  __shared__ float convout[MTILE * CCH];                     // 32 KB, never leaves LDS
  __shared__ int sy[MTILE], sx[MTILE], sq[MTILE];
  __shared__ int s_b, s_map, s_s, s_loc0;

  const int t = threadIdx.x;
  if (t < MTILE) {
    int p = blockIdx.x * MTILE + t;           // tiles never straddle maps (all cums % 32 == 0)
    int b = p / POS_PER_IMG;
    int q = p - b * POS_PER_IMG;
    int map = 0;
    while (map < 4 && q >= d_cum[map + 1]) ++map;
    int s = 128 >> map;
    int loc = q - d_cum[map];
    int y = loc / s;
    sy[t] = y; sx[t] = loc - y * s; sq[t] = q;
    if (t == 0) { s_b = b; s_map = map; s_s = s; s_loc0 = loc; }
  }
  __syncthreads();
  const int b = s_b, map = s_map, s = s_s;
  const int spp = s * s;
  const int base0 = 2048 * d_cum[map] + (b * spp + s_loc0) * CCH;  // halfs, 512B-aligned

  // fixed per-thread staging slots: 1024 x 16B slots; thread owns slot t + j*256
  const int part = t & 31;                    // 16B chunk within a 512B row (constant)
  int my_m[4], my_y[4], my_x[4];
#pragma unroll
  for (int j = 0; j < 4; ++j) {
    my_m[j] = (t >> 5) + j * 8;
    my_y[j] = sy[my_m[j]];
    my_x[j] = sx[my_m[j]];
  }

  const int lane = t & 31, wave = t >> 5;
  const int mloc = lane & 15, hi = lane >> 4;
  const int n0g = wave * 16 + mloc;
  const int n1g = (wave + 8) * 16 + mloc;

  v8f acc00 = {0.f,0.f,0.f,0.f,0.f,0.f,0.f,0.f};
  v8f acc01 = acc00, acc10 = acc00, acc11 = acc00;

#pragma unroll 1
  for (int tap = 0; tap < 9; ++tap) {
    const int ky = tap / 3, kx = tap - (tap / 3) * 3;
    const int shift = (ky - 1) * s + (kx - 1);          // linear position shift of this tap
    // ---- stage tap slab [32 rows x 512B], zero-filled at borders ----
#pragma unroll
    for (int j = 0; j < 4; ++j) {
      int yy = my_y[j] + ky - 1, xx = my_x[j] + kx - 1;
      bool valid = ((unsigned)yy < (unsigned)s) && ((unsigned)xx < (unsigned)s);
      uint4 v; v.x = 0u; v.y = 0u; v.z = 0u; v.w = 0u;
      if (valid) {
        int gidx = base0 + ((my_m[j] + shift) << 8) + (part << 3);   // halfs
        v = *(const uint4*)(fmT + gidx);
      }
      *(uint4*)(smT + my_m[j] * 264 + (part << 3)) = v;
    }
    if (tap < 8) {                                       // prefetch next tap's row
      int ky2 = (tap + 1) / 3, kx2 = (tap + 1) - ((tap + 1) / 3) * 3;
      int sh2 = (ky2 - 1) * s + (kx2 - 1);
      __builtin_prefetch(fmT + base0 + ((my_m[0] + sh2) << 8), 0, 1);
    }
    __syncthreads();
    // ---- 8 WMMA K-steps over this tap's 256 channels ----
    const unsigned short* wb0 = wpre2 + n0g * KTOT + tap * CCH + hi * 8;
    const unsigned short* wb1 = wpre2 + n1g * KTOT + tap * CCH + hi * 8;
#pragma unroll
    for (int c0 = 0; c0 < CCH; c0 += 32) {
      FragBF a0, a1, bf0, bf1;
      a0.q[0] = *(const uint4*)(smT + mloc * 264 + c0 + hi * 8);
      a0.q[1] = *(const uint4*)(smT + mloc * 264 + c0 + hi * 8 + 16);
      a1.q[0] = *(const uint4*)(smT + (16 + mloc) * 264 + c0 + hi * 8);
      a1.q[1] = *(const uint4*)(smT + (16 + mloc) * 264 + c0 + hi * 8 + 16);
      bf0.q[0] = *(const uint4*)(wb0 + c0);
      bf0.q[1] = *(const uint4*)(wb0 + c0 + 16);
      bf1.q[0] = *(const uint4*)(wb1 + c0);
      bf1.q[1] = *(const uint4*)(wb1 + c0 + 16);
      acc00 = wmma_bf16(a0.v, bf0.v, acc00);
      acc01 = wmma_bf16(a0.v, bf1.v, acc01);
      acc10 = wmma_bf16(a1.v, bf0.v, acc10);
      acc11 = wmma_bf16(a1.v, bf1.v, acc11);
    }
    __syncthreads();
  }

  // ---- epilogue: +bias, ReLU, into LDS (C layout: vgpr r -> m = r + hi*8, n = lane&15) ----
  {
    float bp0 = b_pre[n0g];
    float bp1 = b_pre[n1g];
#pragma unroll
    for (int r = 0; r < 8; ++r) {
      int m0 = r + hi * 8, m1 = m0 + 16;
      float v;
      v = acc00[r] + bp0; convout[m0 * CCH + n0g] = v > 0.f ? v : 0.f;
      v = acc01[r] + bp1; convout[m0 * CCH + n1g] = v > 0.f ? v : 0.f;
      v = acc10[r] + bp0; convout[m1 * CCH + n0g] = v > 0.f ? v : 0.f;
      v = acc11[r] + bp1; convout[m1 * CCH + n1g] = v > 0.f ? v : 0.f;
    }
  }
  __syncthreads();

  // ---- 1x1 proj: 32 positions x 9 anchors, 256-deep f32 dot (tiny vs conv) ----
  for (int idx = t; idx < MTILE * NCA; idx += 256) {
    int m = idx / NCA, ai = idx - (idx / NCA) * NCA;
    const float* wp = w_proj + ai * CCH;
    const float* fv = convout + m * CCH;
    float acc = b_proj[ai];
#pragma unroll 4
    for (int c = 0; c < CCH; ++c) acc += fv[c] * wp[c];
    logits[b * ANCH_PER_IMG + sq[m] * NCA + ai] = acc;
  }
}

// ---------------- bitonic sorts (block-wide, LDS) ----------------
__device__ void bitonic_sort_u32(unsigned int* a, int n, int tid, int nth) {
  for (int k = 2; k <= n; k <<= 1)
    for (int j = k >> 1; j > 0; j >>= 1) {
      __syncthreads();
      for (int i = tid; i < n; i += nth) {
        int ixj = i ^ j;
        if (ixj > i) {
          unsigned int x = a[i], y = a[ixj];
          bool up = ((i & k) == 0);
          if ((up && x > y) || (!up && x < y)) { a[i] = y; a[ixj] = x; }
        }
      }
    }
  __syncthreads();
}

__device__ void bitonic_sort_u64(unsigned long long* a, int n, int tid, int nth) {
  for (int k = 2; k <= n; k <<= 1)
    for (int j = k >> 1; j > 0; j >>= 1) {
      __syncthreads();
      for (int i = tid; i < n; i += nth) {
        int ixj = i ^ j;
        if (ixj > i) {
          unsigned long long x = a[i], y = a[ixj];
          bool up = ((i & k) == 0);
          if ((up && x > y) || (!up && x < y)) { a[i] = y; a[ixj] = x; }
        }
      }
    }
  __syncthreads();
}

// ---------------- kernel 3: exact top-300 per image (radix select) ----------------
__global__ __launch_bounds__(1024)
void topk_kernel(const float* __restrict__ logits, int* __restrict__ sel_ids) {
  const int N = ANCH_PER_IMG, NT = 1024;
  const int b = blockIdx.x, tid = threadIdx.x;
  const float* lg = logits + (size_t)b * N;
  __shared__ unsigned int hist[256];
  __shared__ unsigned int sh_prefix; __shared__ int sh_remK;
  __shared__ unsigned long long sArr[512];
  __shared__ unsigned int tieBuf[1024];
  __shared__ unsigned int cntA, cntT;

  unsigned int prefix = 0; int remK = RELK;
  for (int pass = 0; pass < 4; ++pass) {
    int shift = 24 - 8 * pass;
    for (int i = tid; i < 256; i += NT) hist[i] = 0;
    __syncthreads();
    for (int i = tid; i < N; i += NT) {
      unsigned int u = __float_as_uint(lg[i]);
      unsigned int key = u ^ ((u >> 31) ? 0xFFFFFFFFu : 0x80000000u);   // order-preserving
      unsigned int upper = (pass == 0) ? 0u : (key >> (shift + 8));
      if (upper == prefix) atomicAdd(&hist[(key >> shift) & 255u], 1u);
    }
    __syncthreads();
    if (tid == 0) {
      int rk = remK, d = 255;
      for (; d >= 0; --d) { int c = (int)hist[d]; if (rk <= c) break; rk -= c; }
      if (d < 0) d = 0;
      sh_prefix = (prefix << 8) | (unsigned)d;
      sh_remK = rk;
    }
    __syncthreads();
    prefix = sh_prefix; remK = sh_remK;
    __syncthreads();
  }
  const unsigned int T = prefix;        // exact threshold key
  const int tieNeed = remK;
  if (tid == 0) { cntA = 0; cntT = 0; }
  __syncthreads();
  for (int i = tid; i < N; i += NT) {
    unsigned int u = __float_as_uint(lg[i]);
    unsigned int key = u ^ ((u >> 31) ? 0xFFFFFFFFu : 0x80000000u);
    if (key > T) {
      unsigned int p = atomicAdd(&cntA, 1u);
      if (p < 512u) sArr[p] = (((unsigned long long)(~key)) << 32) | (unsigned int)i;
    } else if (key == T) {
      unsigned int p = atomicAdd(&cntT, 1u);
      if (p < 1024u) tieBuf[p] = (unsigned int)i;
    }
  }
  __syncthreads();
  unsigned int nA = cntA < 512u ? cntA : 512u;   // <= 299 by construction
  unsigned int nT = cntT < 1024u ? cntT : 1024u;
  for (int i = tid; i < 1024; i += NT) if ((unsigned)i >= nT) tieBuf[i] = 0xFFFFFFFFu;
  __syncthreads();
  bitonic_sort_u32(tieBuf, 1024, tid, NT);       // smallest tie indices first
  int total = (int)nA + tieNeed;                 // == 300
  if (tid < tieNeed) sArr[nA + tid] = (((unsigned long long)(~T)) << 32) | tieBuf[tid];
  for (int i = tid; i < 512; i += NT) if (i >= total) sArr[i] = ~0ull;
  __syncthreads();
  bitonic_sort_u64(sArr, 512, tid, NT);          // (prob desc, idx asc) via (~key, idx) asc
  if (tid < RELK) sel_ids[b * RELK + tid] = b * N + (int)(unsigned int)(sArr[tid] & 0xFFFFFFFFu);
}

// ---------------- kernel 4: bucket selections by cell anchor ----------------
__global__ __launch_bounds__(256)
void bucket_kernel(const int* __restrict__ sel_ids, int* __restrict__ perm,
                   int* __restrict__ cellOff, int* __restrict__ cellCnt) {
  __shared__ int cnt[NCA]; __shared__ int off[NCA + 1]; __shared__ int cur[NCA];
  int tid = threadIdx.x;
  if (tid < NCA) { cnt[tid] = 0; cur[tid] = 0; }
  __syncthreads();
  const int NSEL = BATCH * RELK;
  for (int n = tid; n < NSEL; n += 256) atomicAdd(&cnt[sel_ids[n] % NCA], 1);
  __syncthreads();
  if (tid == 0) { off[0] = 0; for (int c = 0; c < NCA; ++c) off[c + 1] = off[c] + cnt[c]; }
  __syncthreads();
  for (int n = tid; n < NSEL; n += 256) {
    int c = sel_ids[n] % NCA;
    int p = atomicAdd(&cur[c], 1) + off[c];
    perm[p] = n;                       // order within a cell is irrelevant (scatter by n)
  }
  __syncthreads();
  if (tid < NCA) cellCnt[tid] = cnt[tid];
  if (tid < NCA + 1) cellOff[tid] = off[tid];
}

// ---------------- kernel 5: cell-bucketed gather-GEMM (WMMA) ----------------
// grid = (4 Ntile-groups, 150 Mtiles, 9 cells); block = 128 (4 waves, one 16-wide N tile each)
__global__ __launch_bounds__(128)
void post_gemm_kernel(const unsigned short* __restrict__ fmT,
                      const unsigned short* __restrict__ wpostT,
                      const float* __restrict__ b_post,
                      const int* __restrict__ sel_ids, const int* __restrict__ perm,
                      const int* __restrict__ cellOff, const int* __restrict__ cellCnt,
                      float* __restrict__ out) {
  const int cell = blockIdx.z;
  const int cnt = cellCnt[cell];
  const int mtile = blockIdx.y;
  if (mtile * 16 >= cnt) return;       // uniform early-exit

  __shared__ __align__(16) unsigned short smA[16 * 264];   // 16 rows x 256 ch bf16 (padded)
  __shared__ int rowOut[16];
  __shared__ int rbase[16];
  const int tid = threadIdx.x;

  if (tid < 16) {
    int idx = mtile * 16 + tid;
    int n0 = -1; int id;
    if (idx < cnt) { n0 = perm[cellOff[cell] + idx]; id = sel_ids[n0]; }
    else           { id = sel_ids[perm[cellOff[cell]]]; }   // dummy row, masked at write
    rowOut[tid] = n0;
    int fid = id / NCA;
    int bb = fid / POS_PER_IMG;
    int q  = fid - bb * POS_PER_IMG;
    int map = 0; while (map < 4 && q >= d_cum[map + 1]) ++map;
    int s = 128 >> map;
    int loc = q - d_cum[map];
    rbase[tid] = 2048 * d_cum[map] + (bb * (s * s) + loc) * CCH;   // halfs into fmT
  }
  __syncthreads();

  // gather 16 contiguous 512B feature lines from fmT
#pragma unroll
  for (int j = 0; j < 4; ++j) {
    int slot = tid + j * 128;                // 512 x 16B slots
    int m = slot >> 5, part = slot & 31;
    uint4 v = *(const uint4*)(fmT + rbase[m] + (part << 3));
    *(uint4*)(smA + m * 264 + (part << 3)) = v;
  }
  __syncthreads();

  const int lane = tid & 31, wave = tid >> 5;
  const int ntile = blockIdx.x * 4 + wave;
  const int mloc = lane & 15, hi = lane >> 4;
  const int nglob = ntile * 16 + mloc;
  v8f acc = {0.f, 0.f, 0.f, 0.f, 0.f, 0.f, 0.f, 0.f};
  const unsigned short* wb = wpostT + cell * 65536 + nglob * CCH + hi * 8;
#pragma unroll
  for (int kb = 0; kb < CCH; kb += 32) {
    FragBF a, bf;
    a.q[0] = *(const uint4*)(smA + mloc * 264 + kb + hi * 8);
    a.q[1] = *(const uint4*)(smA + mloc * 264 + kb + hi * 8 + 16);
    bf.q[0] = *(const uint4*)(wb + kb);
    bf.q[1] = *(const uint4*)(wb + kb + 16);
    acc = wmma_bf16(a.v, bf.v, acc);
  }
  float bias = b_post[cell * CCH + nglob];
#pragma unroll
  for (int r = 0; r < 8; ++r) {
    int m = r + hi * 8;
    int n0 = rowOut[m];
    if (n0 >= 0) out[(size_t)n0 * CCH + nglob] = acc[r] + bias;
  }
}

// ---------------- host launcher ----------------
extern "C" void kernel_launch(void* const* d_in, const int* in_sizes, int n_in,
                              void* d_out, int out_size, void* d_ws, size_t ws_size,
                              hipStream_t stream) {
  const float* fm0    = (const float*)d_in[0];
  const float* fm1    = (const float*)d_in[1];
  const float* fm2    = (const float*)d_in[2];
  const float* fm3    = (const float*)d_in[3];
  const float* fm4    = (const float*)d_in[4];
  const float* w_pre  = (const float*)d_in[5];
  const float* b_pre  = (const float*)d_in[6];
  const float* w_proj = (const float*)d_in[7];
  const float* b_proj = (const float*)d_in[8];
  const float* w_post = (const float*)d_in[9];
  const float* b_post = (const float*)d_in[10];

  // workspace layout (bytes, 256-aligned); total ~98 MB
  char* ws = (char*)d_ws;
  float*          logits  = (float*)(ws + 0);                  // 8*196416*4  = 6,285,312
  unsigned short* wpre2   = (unsigned short*)(ws + 6285312);   // 589824*2    = 1,179,648
  unsigned short* wpostT  = (unsigned short*)(ws + 7464960);   // 589824*2    = 1,179,648
  int*            sel_ids = (int*)(ws + 8644608);              // 2400*4
  int*            perm    = (int*)(ws + 8654208);              // 2400*4
  int*            cellOff = (int*)(ws + 8663808);              // 10*4 (padded to 64)
  int*            cellCnt = (int*)(ws + 8663872);              // 9*4  (padded to 192)
  unsigned short* fmT     = (unsigned short*)(ws + 8664064);   // 44,695,552*2 = 89,391,104

  convert_weights_kernel<<<2304, 256, 0, stream>>>(w_pre, w_post, wpre2, wpostT);
  prepass_kernel<<<174592, 256, 0, stream>>>(fm0, fm1, fm2, fm3, fm4, fmT);
  conv_proj_kernel<<<5456, 256, 0, stream>>>(fmT, wpre2, b_pre, w_proj, b_proj, logits);
  topk_kernel<<<8, 1024, 0, stream>>>(logits, sel_ids);
  bucket_kernel<<<1, 256, 0, stream>>>(sel_ids, perm, cellOff, cellCnt);
  post_gemm_kernel<<<dim3(4, 150, 9), 128, 0, stream>>>(fmT, wpostT, b_post, sel_ids, perm,
                                                        cellOff, cellCnt, (float*)d_out);
}
